// Multi_Head_Attention_6966436954399
// MI455X (gfx1250) — matrix operand order repaired and checked by reading, not compiled, not run on hardware
//
#include <hip/hip_runtime.h>

// ---------------------------------------------------------------------------
// Multi-head attention (fp32 in/out) for MI455X / gfx1250.
// Math: V_WMMA_F32_16X16X32_BF16 with bf16x3 split operands (~fp32 accuracy,
// 8x K-depth per WMMA vs f32 16x16x4). Data movement: all tensors pre-split
// once into bf16 hi/lo; hot-loop staging is GLOBAL_LOAD_ASYNC_TO_LDS_B128,
// DOUBLE-BUFFERED so the async DMA overlaps the WMMA pipeline (ASYNCcnt ops
// retire in order, so a partial s_wait_asynccnt is a precise stage fence).
//   D_MODEL=2048, NUM_HEADS=16, HEAD_DIM=128, B=2, S=2048
// ---------------------------------------------------------------------------

typedef __attribute__((ext_vector_type(16))) __bf16 v16bf;
typedef __attribute__((ext_vector_type(8)))  float  v8f;
typedef __attribute__((ext_vector_type(4)))  float  v4f;

static constexpr int BATCH  = 2;
static constexpr int SEQ    = 2048;
static constexpr int DMODEL = 2048;
static constexpr int NHEADS = 16;
static constexpr int HDIM   = 128;
static constexpr int BS     = BATCH * SEQ;   // 4096 rows
static constexpr int D3     = 3 * DMODEL;    // 6144

__device__ __forceinline__ v8f wmma_bf16(v16bf a, v16bf b, v8f c) {
  return __builtin_amdgcn_wmma_f32_16x16x32_bf16(
      false, a, false, b, (short)0, c, false, false);
}

__device__ __forceinline__ v16bf ld_frag(const __bf16* p0, const __bf16* p1) {
  v16bf r;
  ((uint4*)&r)[0] = *(const uint4*)p0;
  ((uint4*)&r)[1] = *(const uint4*)p1;
  return r;
}

__device__ __forceinline__ void split4_store(const v4f v, __bf16* ph, __bf16* pl) {
  union { __bf16 h[4]; uint2 u; } H, L;
#pragma unroll
  for (int j = 0; j < 4; ++j) {
    const float  x  = v[j];
    const __bf16 hh = (__bf16)x;
    H.h[j] = hh;
    L.h[j] = (__bf16)(x - (float)hh);
  }
  *(uint2*)ph = H.u;
  *(uint2*)pl = L.u;
}

// CDNA5 async global->LDS copy, 16B per lane (GV mode), tracked by ASYNCcnt.
__device__ __forceinline__ void async_copy16(void* lds, const void* gptr) {
  asm volatile("global_load_async_to_lds_b128 %0, %1, off"
               :
               : "v"((unsigned)(unsigned long long)lds),
                 "v"((unsigned long long)gptr)
               : "memory");
}
template <int N>
__device__ __forceinline__ void wait_async_le() {
  asm volatile("s_wait_asynccnt %0" :: "i"(N) : "memory");
}

// ---------------------------------------------------------------------------
// One-time fp32 -> bf16 hi/lo split (optionally scaled).
// ---------------------------------------------------------------------------
__global__ __launch_bounds__(256) void split_f32_to_bf16(
    const float* __restrict__ src, __bf16* __restrict__ h,
    __bf16* __restrict__ l, float scale, long long n4)
{
  const long long i = (long long)blockIdx.x * 256 + threadIdx.x;
  if (i >= n4) return;
  const v4f v = *(const v4f*)(src + 4 * i) * scale;
  split4_store(v, h + 4 * i, l + 4 * i);
}

// ---------------------------------------------------------------------------
// C[M,N] = A[M,K] @ W[N,K]^T, pre-split bf16 hi/lo inputs, bf16x3 WMMA.
// 64x64 tile, BK=64, DOUBLE-BUFFERED async-to-LDS staging (8 async/wave/stage).
// splitOut=1: write bf16 hi/lo (Ch/Cl); else fp32 (Cf).
// ---------------------------------------------------------------------------
__global__ __launch_bounds__(256) void gemm_bf16x3(
    const __bf16* __restrict__ Ahg, const __bf16* __restrict__ Alg,
    const __bf16* __restrict__ Bhg, const __bf16* __restrict__ Blg,
    float* __restrict__ Cf, __bf16* __restrict__ Ch, __bf16* __restrict__ Cl,
    int M, int N, int K, int splitOut)
{
  __shared__ __bf16 Ah[2][64][72];   // 144B rows, 16B-aligned chunks
  __shared__ __bf16 Al[2][64][72];
  __shared__ __bf16 Bh[2][64][72];
  __shared__ __bf16 Bl[2][64][72];

  const int tid  = threadIdx.x;
  const int lane = tid & 31;
  const int wave = tid >> 5;
  const int hi   = lane >> 4;
  const int ln   = lane & 15;
  const int wm   = wave & 3;
  const int wn   = wave >> 2;
  const int m0   = blockIdx.y * 64;
  const int n0   = blockIdx.x * 64;

  v8f acc0 = {};
  v8f acc1 = {};

  const __bf16* gbase[4] = { Ahg + (size_t)m0 * K, Alg + (size_t)m0 * K,
                             Bhg + (size_t)n0 * K, Blg + (size_t)n0 * K };

  auto issue = [&](int k0, int bufw) {
    __bf16* lb[4] = { &Ah[bufw][0][0], &Al[bufw][0][0],
                      &Bh[bufw][0][0], &Bl[bufw][0][0] };
#pragma unroll
    for (int i = 0; i < 8; ++i) {
      const int arr   = i >> 1;                // uniform per unrolled i
      const int chunk = tid + (i & 1) * 256;   // 0..511 within array
      const int row   = chunk >> 3;            // 0..63
      const int c8    = (chunk & 7) << 3;      // 0..56
      async_copy16(lb[arr] + row * 72 + c8,
                   gbase[arr] + (size_t)row * K + k0 + c8);
    }
  };

  const int nstage = K >> 6;
  issue(0, 0);
  for (int s = 0; s < nstage; ++s) {
    const int buf = s & 1;
    __syncthreads();                      // buf^1 consumers (stage s-1) done
    if (s + 1 < nstage) {
      issue((s + 1) << 6, buf ^ 1);       // prefetch next stage
      wait_async_le<8>();                 // only next stage's 8 outstanding
    } else {
      wait_async_le<0>();
    }
    __syncthreads();                      // stage s visible to all waves

#pragma unroll
    for (int kk = 0; kk < 64; kk += 32) {
      const __bf16* arh = &Ah[buf][wm * 16 + ln][kk];
      const __bf16* arl = &Al[buf][wm * 16 + ln][kk];
      const v16bf a_h = ld_frag(arh + 16 * hi, arh + 16 * hi + 8);
      const v16bf a_l = ld_frag(arl + 16 * hi, arl + 16 * hi + 8);

      const __bf16* b0h = &Bh[buf][wn * 32 + ln][kk + 16 * hi];
      const __bf16* b0l = &Bl[buf][wn * 32 + ln][kk + 16 * hi];
      const __bf16* b1h = &Bh[buf][wn * 32 + 16 + ln][kk + 16 * hi];
      const __bf16* b1l = &Bl[buf][wn * 32 + 16 + ln][kk + 16 * hi];
      const v16bf B0h = ld_frag(b0h, b0h + 8);
      const v16bf B0l = ld_frag(b0l, b0l + 8);
      const v16bf B1h = ld_frag(b1h, b1h + 8);
      const v16bf B1l = ld_frag(b1l, b1l + 8);

      acc0 = wmma_bf16(a_h, B0h, acc0);
      acc0 = wmma_bf16(a_h, B0l, acc0);
      acc0 = wmma_bf16(a_l, B0h, acc0);
      acc1 = wmma_bf16(a_h, B1h, acc1);
      acc1 = wmma_bf16(a_h, B1l, acc1);
      acc1 = wmma_bf16(a_l, B1h, acc1);
    }
  }

#pragma unroll
  for (int r = 0; r < 8; ++r) {
    const int row = m0 + wm * 16 + r + 8 * hi;
    const size_t base = (size_t)row * N + n0 + wn * 32 + ln;
    if (splitOut) {
      const float y0 = acc0[r];
      const __bf16 h0 = (__bf16)y0;
      Ch[base]      = h0;
      Cl[base]      = (__bf16)(y0 - (float)h0);
      const float y1 = acc1[r];
      const __bf16 h1 = (__bf16)y1;
      Ch[base + 16] = h1;
      Cl[base + 16] = (__bf16)(y1 - (float)h1);
    } else {
      Cf[base]      = acc0[r];
      Cf[base + 16] = acc1[r];
    }
  }
}

// ---------------------------------------------------------------------------
// Causal flash attention on pre-split bf16 qkv[B*S, 6144] (Q pre-scaled).
// Grid (S/128, H, B); 8 waves x 16 q-rows. Double-buffered pipeline:
// K tiles via async-to-LDS (4 async/wave/tile), V tiles via register-staged
// global loads whose LDS transpose-scatter runs after the current compute.
// ---------------------------------------------------------------------------
__global__ __launch_bounds__(256) void flash_attn_bf16x3(
    const __bf16* __restrict__ qkv_h, const __bf16* __restrict__ qkv_l,
    __bf16* __restrict__ attn_h, __bf16* __restrict__ attn_l)
{
  __shared__ __bf16 Ks_h[2][32][136];   // 272B rows
  __shared__ __bf16 Ks_l[2][32][136];
  __shared__ __bf16 Vt_h[2][128][40];   // V transposed, 80B rows
  __shared__ __bf16 Vt_l[2][128][40];
  __shared__ __bf16 Ps_h[8][16][40];    // per-wave P patch
  __shared__ __bf16 Ps_l[8][16][40];

  const int tid  = threadIdx.x;
  const int lane = tid & 31;
  const int wave = tid >> 5;
  const int hi   = lane >> 4;
  const int ln   = lane & 15;

  const int qt = blockIdx.x;
  const int h  = blockIdx.y;
  const int b  = blockIdx.z;

  const int qbase = qt * 128;
  const int qrow0 = qbase + wave * 16;
  const size_t rowbase = (size_t)b * SEQ;

  const int qoff = h * HDIM;
  const int koff = DMODEL + h * HDIM;
  const int voff = 2 * DMODEL + h * HDIM;

  // ---- Q fragments: direct 16B loads (bf16 hi/lo, pre-scaled) ------------
  v16bf qh[4], ql[4];
  {
    const __bf16* qrh = qkv_h + (rowbase + qrow0 + ln) * (size_t)D3 + qoff;
    const __bf16* qrl = qkv_l + (rowbase + qrow0 + ln) * (size_t)D3 + qoff;
#pragma unroll
    for (int kd = 0; kd < 4; ++kd) {
      qh[kd] = ld_frag(qrh + 32 * kd + 16 * hi, qrh + 32 * kd + 16 * hi + 8);
      ql[kd] = ld_frag(qrl + 32 * kd + 16 * hi, qrl + 32 * kd + 16 * hi + 8);
    }
  }

  v8f o[8] = {};
  float rmax[8], rsum[8];
#pragma unroll
  for (int r = 0; r < 8; ++r) { rmax[r] = -3.0e38f; rsum[r] = 0.0f; }

  auto issueK = [&](int kv0, int bufw) {
    __bf16* dsth = &Ks_h[bufw][0][0];
    __bf16* dstl = &Ks_l[bufw][0][0];
#pragma unroll
    for (int i = 0; i < 4; ++i) {
      const int arr   = i >> 1;                // uniform per unrolled i
      const int chunk = tid + (i & 1) * 256;   // 0..511
      const int r     = chunk >> 4;            // 0..31
      const int c8    = (chunk & 15) << 3;     // 0..120
      async_copy16((arr ? dstl : dsth) + r * 136 + c8,
                   (arr ? qkv_l : qkv_h) +
                       (rowbase + kv0 + r) * (size_t)D3 + koff + c8);
    }
  };

  uint4 vstg_h[2], vstg_l[2];                  // V tile in flight
  auto loadV = [&](int kv0) {
#pragma unroll
    for (int i = 0; i < 2; ++i) {
      const int chunk = tid + i * 256;
      const int r     = chunk >> 4;
      const int c8    = (chunk & 15) << 3;
      vstg_h[i] = *(const uint4*)(qkv_h + (rowbase + kv0 + r) * (size_t)D3 + voff + c8);
      vstg_l[i] = *(const uint4*)(qkv_l + (rowbase + kv0 + r) * (size_t)D3 + voff + c8);
    }
  };
  auto scatterV = [&](int bufw) {
#pragma unroll
    for (int i = 0; i < 2; ++i) {
      const int chunk = tid + i * 256;
      const int r     = chunk >> 4;
      const int c8    = (chunk & 15) << 3;
      union { uint4 u; __bf16 e[8]; } Uh, Ul;
      Uh.u = vstg_h[i];
      Ul.u = vstg_l[i];
#pragma unroll
      for (int j = 0; j < 8; ++j) {
        Vt_h[bufw][c8 + j][r] = Uh.e[j];
        Vt_l[bufw][c8 + j][r] = Ul.e[j];
      }
    }
  };

  const int ntiles = (qbase + 128) >> 5;
  issueK(0, 0);
  loadV(0);
  scatterV(0);

  for (int t = 0; t < ntiles; ++t) {
    const int buf = t & 1;
    const int kv0 = t << 5;
    __syncthreads();                     // buf^1 consumers (tile t-1) done
    if (t + 1 < ntiles) {
      issueK((t + 1) << 5, buf ^ 1);     // prefetch next K tile (async DMA)
      loadV((t + 1) << 5);               // prefetch next V tile into regs
      wait_async_le<4>();                // tile t's K complete
    } else {
      wait_async_le<0>();
    }
    __syncthreads();                     // tile t K + V visible

    // ---- S = Q K^T, bf16x3 ----------------------------------------------
    v8f s0 = {};
    v8f s1 = {};
#pragma unroll
    for (int kd = 0; kd < 4; ++kd) {
      const __bf16* k0h = &Ks_h[buf][ln][32 * kd + 16 * hi];
      const __bf16* k0l = &Ks_l[buf][ln][32 * kd + 16 * hi];
      const __bf16* k1h = &Ks_h[buf][16 + ln][32 * kd + 16 * hi];
      const __bf16* k1l = &Ks_l[buf][16 + ln][32 * kd + 16 * hi];
      const v16bf B0h = ld_frag(k0h, k0h + 8);
      const v16bf B0l = ld_frag(k0l, k0l + 8);
      const v16bf B1h = ld_frag(k1h, k1h + 8);
      const v16bf B1l = ld_frag(k1l, k1l + 8);
      s0 = wmma_bf16(qh[kd], B0h, s0);
      s0 = wmma_bf16(qh[kd], B0l, s0);
      s0 = wmma_bf16(ql[kd], B0h, s0);
      s1 = wmma_bf16(qh[kd], B1h, s1);
      s1 = wmma_bf16(qh[kd], B1l, s1);
      s1 = wmma_bf16(ql[kd], B1h, s1);
    }

    // ---- causal mask + online softmax (row = r + 8*hi, col = ln) ---------
#pragma unroll
    for (int r = 0; r < 8; ++r) {
      const int qg = qrow0 + r + 8 * hi;
      if (kv0 + ln > qg)      s0[r] = -3.0e38f;
      if (kv0 + 16 + ln > qg) s1[r] = -3.0e38f;

      float mt = fmaxf(s0[r], s1[r]);
#pragma unroll
      for (int d = 8; d >= 1; d >>= 1) mt = fmaxf(mt, __shfl_xor(mt, d, 16));
      const float mnew = fmaxf(rmax[r], mt);
      const float sc   = __expf(rmax[r] - mnew);
      rmax[r] = mnew;

      const float p0 = __expf(s0[r] - mnew);
      const float p1 = __expf(s1[r] - mnew);
      float ps = p0 + p1;
#pragma unroll
      for (int d = 8; d >= 1; d >>= 1) ps += __shfl_xor(ps, d, 16);
      rsum[r] = rsum[r] * sc + ps;

#pragma unroll
      for (int c = 0; c < 8; ++c) o[c][r] *= sc;

      const __bf16 h0 = (__bf16)p0;
      Ps_h[wave][r + 8 * hi][ln] = h0;
      Ps_l[wave][r + 8 * hi][ln] = (__bf16)(p0 - (float)h0);
      const __bf16 h1 = (__bf16)p1;
      Ps_h[wave][r + 8 * hi][16 + ln] = h1;
      Ps_l[wave][r + 8 * hi][16 + ln] = (__bf16)(p1 - (float)h1);
    }

    // ---- O += P @ V, bf16x3 ----------------------------------------------
    const __bf16* pph = &Ps_h[wave][ln][0];
    const __bf16* ppl = &Ps_l[wave][ln][0];
    const v16bf pah = ld_frag(pph + 16 * hi, pph + 16 * hi + 8);
    const v16bf pal = ld_frag(ppl + 16 * hi, ppl + 16 * hi + 8);
#pragma unroll
    for (int c = 0; c < 8; ++c) {
      const __bf16* vh = &Vt_h[buf][c * 16 + ln][16 * hi];
      const __bf16* vl = &Vt_l[buf][c * 16 + ln][16 * hi];
      const v16bf Bvh = ld_frag(vh, vh + 8);
      const v16bf Bvl = ld_frag(vl, vl + 8);
      o[c] = wmma_bf16(pah, Bvh, o[c]);
      o[c] = wmma_bf16(pah, Bvl, o[c]);
      o[c] = wmma_bf16(pal, Bvh, o[c]);
    }

    // ---- write next V tile into the other buffer (loads long complete) ---
    if (t + 1 < ntiles) scatterV(buf ^ 1);
  }

  // ---- normalize + write attn as bf16 hi/lo ------------------------------
#pragma unroll
  for (int r = 0; r < 8; ++r) {
    const float inv = 1.0f / rsum[r];
    const int row = qrow0 + r + 8 * hi;
    const size_t rb = (rowbase + row) * (size_t)DMODEL + h * HDIM;
#pragma unroll
    for (int c = 0; c < 8; ++c) {
      const float y = o[c][r] * inv;
      const __bf16 hh = (__bf16)y;
      attn_h[rb + c * 16 + ln] = hh;
      attn_l[rb + c * 16 + ln] = (__bf16)(y - (float)hh);
    }
  }
}

// ---------------------------------------------------------------------------
extern "C" void kernel_launch(void* const* d_in, const int* in_sizes, int n_in,
                              void* d_out, int out_size, void* d_ws, size_t ws_size,
                              hipStream_t stream) {
  const float* x      = (const float*)d_in[0];   // [B,S,2048]
  const float* w_qkv  = (const float*)d_in[1];   // [6144,2048]
  const float* w_proj = (const float*)d_in[2];   // [2048,2048]
  float* out = (float*)d_out;                    // [B,S,2048]

  // All-bf16 workspace (~201 MB): hi/lo pairs for every tensor.
  const size_t NW1 = (size_t)D3 * DMODEL;        // 12.58M (Wqkv)
  const size_t NW2 = (size_t)DMODEL * DMODEL;    //  4.19M (Wproj / Wq block)
  const size_t NX  = (size_t)BS * DMODEL;        //  8.39M (x / attn)
  const size_t NQ  = (size_t)BS * D3;            // 25.17M (qkv)

  __bf16* p = (__bf16*)d_ws;
  __bf16* Wqkv_h  = p; p += NW1;
  __bf16* Wqkv_l  = p; p += NW1;
  __bf16* Wproj_h = p; p += NW2;
  __bf16* Wproj_l = p; p += NW2;
  __bf16* x_h     = p; p += NX;
  __bf16* x_l     = p; p += NX;
  __bf16* qkv_h   = p; p += NQ;
  __bf16* qkv_l   = p; p += NQ;
  __bf16* attn_h  = x_h;                         // x dead after GEMM1
  __bf16* attn_l  = x_l;

  const float qscale = 0.08838834764831845f;     // 1/sqrt(128), folded into Wq

  long long n4;
  n4 = (long long)(NX / 4);
  split_f32_to_bf16<<<dim3((unsigned)((n4 + 255) / 256)), 256, 0, stream>>>(
      x, x_h, x_l, 1.0f, n4);
  n4 = (long long)(NW2 / 4);                     // Wq rows 0..2047, pre-scaled
  split_f32_to_bf16<<<dim3((unsigned)((n4 + 255) / 256)), 256, 0, stream>>>(
      w_qkv, Wqkv_h, Wqkv_l, qscale, n4);
  n4 = (long long)((NW1 - NW2) / 4);             // Wk, Wv rows
  split_f32_to_bf16<<<dim3((unsigned)((n4 + 255) / 256)), 256, 0, stream>>>(
      w_qkv + NW2, Wqkv_h + NW2, Wqkv_l + NW2, 1.0f, n4);
  n4 = (long long)(NW2 / 4);
  split_f32_to_bf16<<<dim3((unsigned)((n4 + 255) / 256)), 256, 0, stream>>>(
      w_proj, Wproj_h, Wproj_l, 1.0f, n4);

  // qkv = x @ Wqkv^T   (split bf16 output)
  gemm_bf16x3<<<dim3(D3 / 64, BS / 64), 256, 0, stream>>>(
      x_h, x_l, Wqkv_h, Wqkv_l, nullptr, qkv_h, qkv_l, BS, D3, DMODEL, 1);

  flash_attn_bf16x3<<<dim3(SEQ / 128, NHEADS, BATCH), 256, 0, stream>>>(
      qkv_h, qkv_l, attn_h, attn_l);

  // out = attn @ Wproj^T  (fp32 output)
  gemm_bf16x3<<<dim3(DMODEL / 64, BS / 64), 256, 0, stream>>>(
      attn_h, attn_l, Wproj_h, Wproj_l, out, nullptr, nullptr,
      BS, DMODEL, DMODEL, 0);
}
